// ATTNCritic_76836964925909
// MI455X (gfx1250) — matrix operand order repaired and checked by reading, not compiled, not run on hardware
//
#include <hip/hip_runtime.h>
#include <hip/hip_bf16.h>

// ---------------------------------------------------------------- constants
#define BS_      8
#define T_       256
#define A_       160     // N_AGENTS * N_ACTIONS
#define E_       128
#define D_       128
#define HID_     32
#define S_       256
#define BT_      2048    // BS*T
#define G3_      384     // 3*E
#define AE_      288     // A + E
#define EMBED_   9216    // HID*(A+E)

typedef __attribute__((ext_vector_type(16))) _Float16 v16h;
typedef __attribute__((ext_vector_type(8)))  float    v8f;
typedef __attribute__((ext_vector_type(4)))  unsigned v4u;
typedef __attribute__((ext_vector_type(8)))  unsigned v8u;

union Frag16 { v16h v; unsigned u[8]; };

// ---------------------------------------------------------------- K0: GI = shift(act) @ Wih^T + bih
__global__ __launch_bounds__(384) void gi_kernel(const float* __restrict__ act,
                                                 const float* __restrict__ wih,
                                                 const float* __restrict__ bih,
                                                 float* __restrict__ gi) {
  __shared__ float xr[A_];
  int bt = blockIdx.x;
  int t  = bt & (T_ - 1);
  int j  = threadIdx.x;
  if (j < A_) xr[j] = (t == 0) ? 0.f : act[(size_t)(bt - 1) * A_ + j];
  __syncthreads();
  float acc = bih[j];
  const float* wr = &wih[(size_t)j * A_];
  #pragma unroll 8
  for (int k = 0; k < A_; ++k) acc = fmaf(xr[k], wr[k], acc);
  gi[(size_t)bt * G3_ + j] = acc;
}

// ---------------------------------------------------------------- K1: sequential GRU scan (one block per batch)
__global__ __launch_bounds__(384) void gru_scan_kernel(const float* __restrict__ gi,
                                                       const float* __restrict__ whh,
                                                       const float* __restrict__ bhh,
                                                       float* __restrict__ enc) {
  __shared__ float h[E_];
  __shared__ float gh[G3_];
  int b = blockIdx.x, j = threadIdx.x;
  if (j < E_) h[j] = 0.f;
  __syncthreads();
  const float* wrow = &whh[(size_t)j * E_];
  for (int t = 0; t < T_; ++t) {
    float acc = bhh[j];
    #pragma unroll 8
    for (int k = 0; k < E_; ++k) acc = fmaf(h[k], wrow[k], acc);
    gh[j] = acc;
    __syncthreads();                        // gh ready, all reads of old h done
    if (j < E_) {
      size_t base = ((size_t)b * T_ + t) * G3_;
      float ir = gi[base + j], iz = gi[base + E_ + j], in_ = gi[base + 2 * E_ + j];
      float r = 1.f / (1.f + __expf(-(ir + gh[j])));
      float z = 1.f / (1.f + __expf(-(iz + gh[E_ + j])));
      float n = tanhf(in_ + r * gh[2 * E_ + j]);
      float hn = (1.f - z) * n + z * h[j];
      h[j] = hn;
      enc[((size_t)b * T_ + t) * E_ + j] = hn;
    }
    __syncthreads();
  }
}

// ---------------------------------------------------------------- K2: s0 = tanh(enc@fc^T+b); ps = s0@Ws^T; pe = enc@We^T
__global__ __launch_bounds__(128) void post_enc_kernel(const float* __restrict__ enc,
                                                       const float* __restrict__ fc_w,
                                                       const float* __restrict__ fc_b,
                                                       const float* __restrict__ attn_w,
                                                       float* __restrict__ ps,
                                                       float* __restrict__ pe) {
  __shared__ float er[D_], s0r[D_];
  int bt = blockIdx.x, d = threadIdx.x;
  er[d] = enc[(size_t)bt * E_ + d];
  __syncthreads();
  float a0 = fc_b[d];
  const float* fw = &fc_w[(size_t)d * E_];
  #pragma unroll 8
  for (int k = 0; k < E_; ++k) a0 = fmaf(er[k], fw[k], a0);
  s0r[d] = tanhf(a0);
  float pev = 0.f;
  const float* we = &attn_w[(size_t)d * (D_ + E_) + D_];
  #pragma unroll 8
  for (int k = 0; k < E_; ++k) pev = fmaf(er[k], we[k], pev);
  pe[(size_t)bt * D_ + d] = pev;
  __syncthreads();
  float psv = 0.f;
  const float* wsr = &attn_w[(size_t)d * (D_ + E_)];
  #pragma unroll 8
  for (int k = 0; k < D_; ++k) psv = fmaf(s0r[k], wsr[k], psv);
  ps[(size_t)bt * D_ + d] = psv;
}

// ---------------------------------------------------------------- K3: causal additive attention + build inp = [act, c]
__global__ __launch_bounds__(256) void attn_kernel(const float* __restrict__ ps,
                                                   const float* __restrict__ pe,
                                                   const float* __restrict__ vw,
                                                   const float* __restrict__ enc,
                                                   const float* __restrict__ act,
                                                   float* __restrict__ inp) {
  __shared__ float psr[D_], vws[D_], sc[T_], red[T_];
  int bq = blockIdx.x;
  int b = bq >> 8, q = bq & (T_ - 1);
  int k = threadIdx.x;
  if (k < D_) { psr[k] = ps[(size_t)bq * D_ + k]; vws[k] = vw[k]; }
  __syncthreads();
  float s = -INFINITY;
  if (k <= q) {
    s = 0.f;
    const float* per = &pe[((size_t)b * T_ + k) * D_];
    #pragma unroll 4
    for (int d = 0; d < D_; ++d) s = fmaf(vws[d], tanhf(psr[d] + per[d]), s);
  }
  red[k] = s;
  __syncthreads();
  for (int off = 128; off > 0; off >>= 1) {
    if (k < off) red[k] = fmaxf(red[k], red[k + off]);
    __syncthreads();
  }
  float mx = red[0];
  __syncthreads();
  float e = (k <= q) ? __expf(s - mx) : 0.f;
  red[k] = e;
  __syncthreads();
  for (int off = 128; off > 0; off >>= 1) {
    if (k < off) red[k] += red[k + off];
    __syncthreads();
  }
  float inv = 1.f / red[0];
  sc[k] = e * inv;
  __syncthreads();
  if (k < E_) {
    float acc = 0.f;
    for (int kk = 0; kk <= q; ++kk)
      acc = fmaf(sc[kk], enc[((size_t)b * T_ + kk) * E_ + k], acc);
    inp[(size_t)bq * AE_ + A_ + k] = acc;
  }
  if (k < A_) inp[(size_t)bq * AE_ + k] = act[(size_t)bq * A_ + k];
}

// ---------------------------------------------------------------- K4/K5: WMMA GEMM, BM=128 BN=128 BK=32, 8 waves (4x2, 32x64 per wave)
// EPI==0 : C = relu(A_f32@B^T + bias)  -> f16 store      (r1 = relu(st@hw1_w0^T+b0))
// EPI==1 : C = A_f16@B^T + bias, fused hpre[b,m%32] += inp[b,m/32]*C  (never materialize w1)
//          A tile staged by the Tensor Data Mover (tensor_load_to_lds), padded in-flight
//          to the bank-conflict-free As[128][40] layout via D# pad fields.
template<int EPI>
__global__ __launch_bounds__(256) void hyper_gemm(const float*    __restrict__ Af32,
                                                  const _Float16* __restrict__ Ah16,
                                                  const float*    __restrict__ Bw,
                                                  const float*    __restrict__ biasN,
                                                  _Float16*       __restrict__ outH,
                                                  const float*    __restrict__ inpv,
                                                  float*          __restrict__ hpre,
                                                  int K) {
  __shared__ alignas(16) _Float16 As[128][40];    // 64B data + 16B pad per row
  __shared__ alignas(16) _Float16 Bs[32][136];
  __shared__ float hacc[EPI ? 128 : 1][HID_];

  const int rowBase = blockIdx.x * 128;   // b-tiles fastest -> col tile of W shared in L2
  const int colBase = blockIdx.y * 128;
  const int lane = threadIdx.x & 31;
  const int w    = threadIdx.x >> 5;
  const int wr   = w >> 1;                // 0..3 -> 32-row strip
  const int wc   = w & 1;                 // 0..1 -> 64-col strip

  if (EPI == 1) {
    for (int idx = threadIdx.x; idx < 128 * HID_; idx += 256) (&hacc[0][0])[idx] = 0.f;
  }

  v8f acc[2][4];
  #pragma unroll
  for (int rt = 0; rt < 2; ++rt)
    #pragma unroll
    for (int nt = 0; nt < 4; ++nt)
      acc[rt][nt] = (v8f){0, 0, 0, 0, 0, 0, 0, 0};

  for (int k0 = 0; k0 < K; k0 += 32) {
    // ---- stage A tile (128 rows x 32 halves)
    if (EPI == 1) {
      // Tensor Data Mover: one wave-uniform descriptor, issued by a single wave.
      if (threadIdx.x == 0) {
        unsigned lds_base = (unsigned)(uintptr_t)(&As[0][0]);
        unsigned long long ga =
            (unsigned long long)(uintptr_t)(Ah16 + (size_t)rowBase * K + k0);
        v4u g0;
        g0.x = 1u;                                  // count=1, user mode
        g0.y = lds_base;                            // lds_addr
        g0.z = (unsigned)ga;                        // global_addr[31:0]
        g0.w = (unsigned)(ga >> 32) | (2u << 30);   // global_addr[56:32] | type=2
        v8u g1;
        // data_size=2B(code1), pad_enable, pad_interval=16DW(code3), pad_amount=4DW(code3)
        g1.s0 = (1u << 16) | (1u << 20) | (3u << 22) | (3u << 25);
        unsigned td0 = (unsigned)K;                 // tensor_dim0 (halves)
        unsigned td1 = 1u << 20;                    // tensor_dim1 (large, tiles never OOB)
        g1.s1 = (td0 & 0xFFFFu) << 16;              // abar_addr=0 | td0.lo
        g1.s2 = (td0 >> 16) | ((td1 & 0xFFFFu) << 16);
        g1.s3 = (td1 >> 16) | (32u << 16);          // td1.hi | tile_dim0=32
        g1.s4 = 128u;                               // tile_dim1=128 | tile_dim2=0
        g1.s5 = (unsigned)K;                        // tensor_dim0_stride.lo (halves)
        g1.s6 = 0u;                                 // stride0.hi | stride1.lo
        g1.s7 = 0u;
        asm volatile("tensor_load_to_lds %0, %1" :: "s"(g0), "s"(g1) : "memory");
        __builtin_amdgcn_s_wait_tensorcnt(0);
      }
    } else {
      int row = threadIdx.x >> 1;
      int off = (threadIdx.x & 1) * 16;
      const float* src = &Af32[(size_t)(rowBase + row) * K + k0 + off];
      alignas(16) _Float16 tmp[16];
      #pragma unroll
      for (int i = 0; i < 16; ++i) tmp[i] = (_Float16)src[i];
      *(uint4*)&As[row][off]     = ((const uint4*)tmp)[0];
      *(uint4*)&As[row][off + 8] = ((const uint4*)tmp)[1];
    }
    { // ---- stage B tile: Bs[k][n] = W[colBase+n][k0+k] (fp32 -> packed f16 pairs)
      int npair = threadIdx.x & 63;       // n = 2*npair
      int kg    = threadIdx.x >> 6;       // koff = kg*8
      int n     = npair * 2;
      const float* s0p = &Bw[(size_t)(colBase + n) * K + k0 + kg * 8];
      const float* s1p = s0p + K;
      if (k0 + 32 < K) __builtin_prefetch(s0p + 32, 0, 0);   // global_prefetch next BK
      #pragma unroll
      for (int i = 0; i < 8; ++i) {
        union { unsigned u; _Float16 h[2]; } pk;
        pk.h[0] = (_Float16)s0p[i];
        pk.h[1] = (_Float16)s1p[i];
        *(unsigned*)&Bs[kg * 8 + i][n] = pk.u;
      }
    }
    __syncthreads();

    // build fragments per ISA 16-bit A 16x32 / B 32x16 wave32 layouts
    Frag16 a[2], bfr[4];
    const int ml = lane & 15, g = lane >> 4;
    #pragma unroll
    for (int rt = 0; rt < 2; ++rt) {
      int row = wr * 32 + rt * 16 + ml;
      #pragma unroll
      for (int v = 0; v < 8; ++v) {
        int klo = (v < 4 ? 2 * v : 16 + 2 * (v - 4)) + 8 * g;
        a[rt].u[v] = *(const unsigned*)&As[row][klo];
      }
    }
    #pragma unroll
    for (int nt = 0; nt < 4; ++nt) {
      int nb = wc * 64 + nt * 16;
      #pragma unroll
      for (int s = 0; s < 16; ++s)
        bfr[nt].v[s] = Bs[(s < 8 ? s : 8 + s) + 8 * g][nb + ml];
    }
    #pragma unroll
    for (int rt = 0; rt < 2; ++rt)
      #pragma unroll
      for (int nt = 0; nt < 4; ++nt)
        acc[rt][nt] = __builtin_amdgcn_wmma_f32_16x16x32_f16(
            false, a[rt].v, false, bfr[nt].v, (short)0, acc[rt][nt], false, false);
    __syncthreads();
  }

  // epilogue: C layout — lane l, vgpr r: M = r + 8*(l/16), N = l%16
  const int nl  = lane & 15;
  const int hiM = (lane >> 4) * 8;
  #pragma unroll
  for (int rt = 0; rt < 2; ++rt) {
    #pragma unroll
    for (int nt = 0; nt < 4; ++nt) {
      #pragma unroll
      for (int r = 0; r < 8; ++r) {
        int Mloc = wr * 32 + rt * 16 + hiM + r;
        int j    = colBase + wc * 64 + nt * 16 + nl;
        float val = acc[rt][nt][r] + biasN[j];
        if (EPI == 0) {
          val = fmaxf(val, 0.f);
          outH[(size_t)(rowBase + Mloc) * EMBED_ + j] = (_Float16)val;
        } else {
          atomicAdd(&hacc[Mloc][j & (HID_ - 1)],
                    inpv[(size_t)(rowBase + Mloc) * AE_ + (j >> 5)] * val);
        }
      }
    }
  }
  if (EPI == 1) {
    __syncthreads();
    for (int idx = threadIdx.x; idx < 128 * HID_; idx += 256) {
      int rl = idx >> 5, hh = idx & (HID_ - 1);
      atomicAdd(&hpre[(size_t)(rowBase + rl) * HID_ + hh], hacc[rl][hh]);
    }
  }
}

// ---------------------------------------------------------------- K6: small hyper-heads + final q
__global__ __launch_bounds__(128) void final_kernel(const float* __restrict__ st,
                                                    const float* __restrict__ hwf_w0, const float* __restrict__ hwf_b0,
                                                    const float* __restrict__ hwf_w1, const float* __restrict__ hwf_b1,
                                                    const float* __restrict__ hb1_w,  const float* __restrict__ hb1_b,
                                                    const float* __restrict__ hb2_w0, const float* __restrict__ hb2_b0,
                                                    const float* __restrict__ hb2_w1, const float* __restrict__ hb2_b1,
                                                    const float* __restrict__ hpre,
                                                    float* __restrict__ out) {
  __shared__ float str[S_];
  __shared__ float t1[HID_], tb[HID_], b1v[HID_], prod[HID_], pb[HID_];
  int bt = blockIdx.x, tid = threadIdx.x;
  str[tid]       = st[(size_t)bt * S_ + tid];
  str[tid + 128] = st[(size_t)bt * S_ + tid + 128];
  __syncthreads();
  if (tid < 96) {
    int hh = tid & (HID_ - 1), grp = tid >> 5;
    const float* wrow = (grp == 0 ? hwf_w0 : (grp == 1 ? hb2_w0 : hb1_w)) + (size_t)hh * S_;
    float acc = (grp == 0 ? hwf_b0[hh] : (grp == 1 ? hb2_b0[hh] : hb1_b[hh]));
    #pragma unroll 8
    for (int kk = 0; kk < S_; ++kk) acc = fmaf(str[kk], wrow[kk], acc);
    if (grp == 0)      t1[hh]  = fmaxf(acc, 0.f);
    else if (grp == 1) tb[hh]  = fmaxf(acc, 0.f);
    else               b1v[hh] = acc;
  }
  __syncthreads();
  if (tid < HID_) {
    float wf = hwf_b1[tid];
    const float* w1r = &hwf_w1[(size_t)tid * HID_];
    #pragma unroll
    for (int jj = 0; jj < HID_; ++jj) wf = fmaf(t1[jj], w1r[jj], wf);
    float hval = fmaxf(hpre[(size_t)bt * HID_ + tid] + b1v[tid], 0.f);
    prod[tid] = hval * wf;
    pb[tid]   = tb[tid] * hb2_w1[tid];
  }
  __syncthreads();
  if (tid == 0) {
    float qv = hb2_b1[0];
    #pragma unroll
    for (int jj = 0; jj < HID_; ++jj) qv += prod[jj] + pb[jj];
    out[bt] = qv;
  }
}

// ---------------------------------------------------------------- launch
extern "C" void kernel_launch(void* const* d_in, const int* in_sizes, int n_in,
                              void* d_out, int out_size, void* d_ws, size_t ws_size,
                              hipStream_t stream) {
  const float* act     = (const float*)d_in[0];
  const float* states  = (const float*)d_in[1];
  const float* gru_wih = (const float*)d_in[2];
  const float* gru_whh = (const float*)d_in[3];
  const float* gru_bih = (const float*)d_in[4];
  const float* gru_bhh = (const float*)d_in[5];
  const float* fc_w    = (const float*)d_in[6];
  const float* fc_b    = (const float*)d_in[7];
  const float* attn_w  = (const float*)d_in[8];
  const float* v_w     = (const float*)d_in[9];
  const float* hw1_w0  = (const float*)d_in[10];
  const float* hw1_b0  = (const float*)d_in[11];
  const float* hw1_w1  = (const float*)d_in[12];
  const float* hw1_b1  = (const float*)d_in[13];
  const float* hwf_w0  = (const float*)d_in[14];
  const float* hwf_b0  = (const float*)d_in[15];
  const float* hwf_w1  = (const float*)d_in[16];
  const float* hwf_b1  = (const float*)d_in[17];
  const float* hb1_w   = (const float*)d_in[18];
  const float* hb1_b   = (const float*)d_in[19];
  const float* hb2_w0  = (const float*)d_in[20];
  const float* hb2_b0  = (const float*)d_in[21];
  const float* hb2_w1  = (const float*)d_in[22];
  const float* hb2_b1  = (const float*)d_in[23];

  char* ws = (char*)d_ws;
  size_t off = 0;
  auto carve = [&](size_t bytes) -> void* {
    void* p = ws + off;
    off = (off + bytes + 255) & ~(size_t)255;
    return p;
  };
  float*    gi   = (float*)   carve((size_t)BT_ * G3_ * sizeof(float));
  float*    enc  = (float*)   carve((size_t)BT_ * E_ * sizeof(float));
  float*    ps   = (float*)   carve((size_t)BT_ * D_ * sizeof(float));
  float*    pe   = (float*)   carve((size_t)BT_ * D_ * sizeof(float));
  float*    inp  = (float*)   carve((size_t)BT_ * AE_ * sizeof(float));
  _Float16* r1   = (_Float16*)carve((size_t)BT_ * EMBED_ * sizeof(_Float16));
  float*    hpre = (float*)   carve((size_t)BT_ * HID_ * sizeof(float));

  gi_kernel<<<dim3(BT_), dim3(G3_), 0, stream>>>(act, gru_wih, gru_bih, gi);
  gru_scan_kernel<<<dim3(BS_), dim3(G3_), 0, stream>>>(gi, gru_whh, gru_bhh, enc);
  post_enc_kernel<<<dim3(BT_), dim3(D_), 0, stream>>>(enc, fc_w, fc_b, attn_w, ps, pe);
  attn_kernel<<<dim3(BT_), dim3(T_), 0, stream>>>(ps, pe, v_w, enc, act, inp);

  // r1 = relu(st @ hw1_w0^T + b0), stored f16 (36 MB -> L2-resident for the big GEMM)
  hyper_gemm<0><<<dim3(BT_ / 128, EMBED_ / 128), dim3(256), 0, stream>>>(
      states, nullptr, hw1_w0, hw1_b0, r1, nullptr, nullptr, S_);

  hipMemsetAsync(hpre, 0, (size_t)BT_ * HID_ * sizeof(float), stream);

  // fused 2048x9216x9216 GEMM + inp contraction (348 GFLOP, the whole job)
  hyper_gemm<1><<<dim3(BT_ / 128, EMBED_ / 128), dim3(256), 0, stream>>>(
      nullptr, r1, hw1_w1, hw1_b1, nullptr, inp, hpre, EMBED_);

  final_kernel<<<dim3(BT_), dim3(128), 0, stream>>>(
      states, hwf_w0, hwf_b0, hwf_w1, hwf_b1, hb1_w, hb1_b,
      hb2_w0, hb2_b0, hb2_w1, hb2_b1, hpre, (float*)d_out);

  (void)in_sizes; (void)n_in; (void)out_size; (void)ws_size;
}
